// EdgeAwareGIN_16174846836940
// MI455X (gfx1250) — matrix operand order repaired and checked by reading, not compiled, not run on hardware
//
#include <hip/hip_runtime.h>
#include <cstdint>
#include <cstddef>

typedef __attribute__((ext_vector_type(16))) _Float16 v16h;
typedef __attribute__((ext_vector_type(8)))  _Float16 v8h;
typedef __attribute__((ext_vector_type(8)))  float    v8f;

namespace {

constexpr int kN   = 50000;
constexpr int kE   = 800000;
constexpr int kB   = 512;
constexpr int kH   = 256;
constexpr int kOUT = 512;

// ---------------- elementwise utility kernels ----------------
__global__ __launch_bounds__(256) void zero_f32(float* __restrict__ p, size_t n) {
  size_t i = (size_t)blockIdx.x * blockDim.x + threadIdx.x;
  if (i < n) p[i] = 0.0f;
}

__global__ __launch_bounds__(256) void f32_to_f16(const float* __restrict__ a,
                                                  _Float16* __restrict__ b, size_t n) {
  size_t i = (size_t)blockIdx.x * blockDim.x + threadIdx.x;
  if (i < n) b[i] = (_Float16)a[i];
}

// z = h + agg ; also emit f16 copy of z (next GEMM's A operand)
__global__ __launch_bounds__(256) void add_convert(const float* __restrict__ h,
                                                   const float* __restrict__ agg,
                                                   float* __restrict__ z,
                                                   _Float16* __restrict__ zh, size_t n) {
  size_t i = (size_t)blockIdx.x * blockDim.x + threadIdx.x;
  if (i < n) { float v = h[i] + agg[i]; z[i] = v; zh[i] = (_Float16)v; }
}

// Wt[n*K + k] = (f16) W[k*Nc + n]   (transpose + convert so B frags are contiguous)
__global__ __launch_bounds__(256) void wxpose(const float* __restrict__ W,
                                              _Float16* __restrict__ Wt, int K, int Nc) {
  int i = blockIdx.x * blockDim.x + threadIdx.x;
  if (i >= K * Nc) return;
  int n = i / K, k = i - n * K;
  Wt[i] = (_Float16)W[(size_t)k * Nc + n];
}

struct Tabs9 { const float* t[9]; int d[9]; };
struct Tabs3 { const float* t[3]; int d[3]; };

__global__ __launch_bounds__(256) void embed_node(const int* __restrict__ x, Tabs9 tp,
                                                  _Float16* __restrict__ emb, int N) {
  int i = blockIdx.x * blockDim.x + threadIdx.x;
  if (i >= N * 576) return;
  int n = i / 576, cc = i - n * 576;
  int f = cc >> 6, c = cc & 63;
  int idx = x[n * 9 + f];
  int d = tp.d[f];
  idx = idx < 0 ? 0 : (idx >= d ? d - 1 : idx);
  emb[i] = (_Float16)tp.t[f][idx * 64 + c];
}

__global__ __launch_bounds__(256) void embed_edge(const int* __restrict__ ea, Tabs3 tp,
                                                  _Float16* __restrict__ emb, int E) {
  int i = blockIdx.x * blockDim.x + threadIdx.x;
  if (i >= E * 96) return;
  int e = i / 96, cc = i - e * 96;
  int f = cc >> 5, c = cc & 31;
  int idx = ea[e * 3 + f];
  int d = tp.d[f];
  idx = idx < 0 ? 0 : (idx >= d ? d - 1 : idx);
  emb[i] = (_Float16)tp.t[f][idx * 32 + c];
}

// msg = relu(h[src] + edge_h) ; agg[dst] += msg.  32 lanes/edge, 8 channels/lane.
__global__ __launch_bounds__(256) void msg_agg(const int* __restrict__ ei,
                                               const _Float16* __restrict__ eh,
                                               const float* __restrict__ h,
                                               float* __restrict__ agg, int E) {
  int t = blockIdx.x * blockDim.x + threadIdx.x;
  int e = t >> 5;
  if (e >= E) return;
  int c0 = (t & 31) * 8;
  int src = ei[e];
  int dst = ei[E + e];
  const float* hs = h + (size_t)src * kH + c0;
  v8h ev = *(const v8h*)(eh + (size_t)e * kH + c0);
  float* ag = agg + (size_t)dst * kH + c0;
  float4 h0 = *(const float4*)hs;
  float4 h1 = *(const float4*)(hs + 4);
  float m[8] = {h0.x, h0.y, h0.z, h0.w, h1.x, h1.y, h1.z, h1.w};
#pragma unroll
  for (int i = 0; i < 8; i++) {
    float v = m[i] + (float)ev[i];
    v = v > 0.0f ? v : 0.0f;
    atomicAdd(ag + i, v);
  }
}

// h += relu(layernorm(z) * g + beta)   -- one wave32 per node, 8 ch/lane
__global__ __launch_bounds__(256) void ln_relu_res(const float* __restrict__ z,
                                                   const float* __restrict__ g,
                                                   const float* __restrict__ beta,
                                                   float* __restrict__ h, int N) {
  int wid = (blockIdx.x * blockDim.x + threadIdx.x) >> 5;
  int lane = threadIdx.x & 31;
  if (wid >= N) return;
  const float* zp = z + (size_t)wid * kH;
  float vals[8];
  float s = 0.0f;
#pragma unroll
  for (int i = 0; i < 8; i++) { vals[i] = zp[lane + i * 32]; s += vals[i]; }
#pragma unroll
  for (int off = 16; off > 0; off >>= 1) s += __shfl_xor(s, off, 32);
  float mu = s * (1.0f / 256.0f);
  float vs = 0.0f;
#pragma unroll
  for (int i = 0; i < 8; i++) { float d = vals[i] - mu; vs += d * d; }
#pragma unroll
  for (int off = 16; off > 0; off >>= 1) vs += __shfl_xor(vs, off, 32);
  float r = rsqrtf(vs * (1.0f / 256.0f) + 1e-5f);
  float* hp = h + (size_t)wid * kH;
#pragma unroll
  for (int i = 0; i < 8; i++) {
    int c = lane + i * 32;
    float y = (vals[i] - mu) * r * g[c] + beta[c];
    y = y > 0.0f ? y : 0.0f;
    hp[c] += y;
  }
}

__global__ __launch_bounds__(256) void count_nodes(const int* __restrict__ bids,
                                                   float* __restrict__ counts, int N) {
  int i = blockIdx.x * blockDim.x + threadIdx.x;
  if (i < N) atomicAdd(&counts[bids[i]], 1.0f);
}

__global__ __launch_bounds__(256) void pool_div(float* __restrict__ out,
                                                const float* __restrict__ counts, int total) {
  int i = blockIdx.x * blockDim.x + threadIdx.x;
  if (i >= total) return;
  float c = counts[i / kOUT];
  out[i] /= (c > 1.0f ? c : 1.0f);
}

// ---------------- WMMA GEMM: C[M,Nc] = A[M,K] * Bt[Nc,K]^T + bias ----------------
// 128 threads = 4 wave32; block tile 128x64; wave tile 32x64 (2x4 wmma accums); K step 32.
// Software-pipelined fill: all 6 global b128 loads issued before the barrier,
// overlapping the previous iteration's WMMA chain.
enum { EPI_F32 = 0, EPI_F16 = 1, EPI_F16_RELU = 2, EPI_POOL = 3 };

template <int EPI>
__global__ __launch_bounds__(128) void gemm16(const _Float16* __restrict__ A,
                                              const _Float16* __restrict__ Bt,
                                              const float* __restrict__ bias,
                                              float* __restrict__ Cf,
                                              _Float16* __restrict__ Ch,
                                              const int* __restrict__ bids,
                                              float* __restrict__ pool,
                                              int M, int K, int Nc) {
  __shared__ __align__(32) _Float16 As[128 * 32];
  __shared__ __align__(32) _Float16 Bs[64 * 32];
  const int tid  = threadIdx.x;
  const int lane = tid & 31;
  const int wave = tid >> 5;
  const int m0 = blockIdx.x * 128;
  const int n0 = blockIdx.y * 64;

  // per-thread chunk coordinates (fixed across K iterations)
  int arow[4], asub[4];
#pragma unroll
  for (int i = 0; i < 4; i++) {
    int c = tid + i * 128;
    arow[i] = c >> 2; asub[i] = c & 3;
  }
  // clamp rows once (out-of-range rows masked at epilogue; clamped reads stay in-bounds)
  int agrow[4];
#pragma unroll
  for (int i = 0; i < 4; i++) {
    int gm = m0 + arow[i];
    agrow[i] = gm < M ? gm : M - 1;
  }

  v8f acc[2][4];
#pragma unroll
  for (int mi = 0; mi < 2; mi++)
#pragma unroll
    for (int j = 0; j < 4; j++) acc[mi][j] = {};

  for (int k0 = 0; k0 < K; k0 += 32) {
    // stage all global loads first (pipelined against previous WMMAs)
    uint4 ra[4], rb[2];
#pragma unroll
    for (int i = 0; i < 4; i++)
      ra[i] = *(const uint4*)(A + (size_t)agrow[i] * K + k0 + asub[i] * 8);
#pragma unroll
    for (int i = 0; i < 2; i++)
      rb[i] = *(const uint4*)(Bt + (size_t)(n0 + arow[i]) * K + k0 + asub[i] * 8);
    if (k0 + 32 < K) {               // hint: prefetch next A tile rows
      int gm = m0 + tid;
      if (gm >= M) gm = M - 1;
      __builtin_prefetch(A + (size_t)gm * K + k0 + 32, 0, 1);
    }
    __syncthreads();                 // previous iteration's fragment reads done
#pragma unroll
    for (int i = 0; i < 4; i++)
      *(uint4*)(As + arow[i] * 32 + asub[i] * 8) = ra[i];
#pragma unroll
    for (int i = 0; i < 2; i++)
      *(uint4*)(Bs + arow[i] * 32 + asub[i] * 8) = rb[i];
    __syncthreads();

    // A fragments (16x32 f16): lane<16 -> row l, K[0..15]; lane>=16 -> K[16..31]
    const int klo = (lane >> 4) * 16;
    v16h afrag[2];
#pragma unroll
    for (int mi = 0; mi < 2; mi++) {
      const int ar = wave * 32 + mi * 16 + (lane & 15);
      afrag[mi] = *(const v16h*)(As + ar * 32 + klo);
    }
    // B fragments (32x16 f16): lane<16 -> col, K[0..15]; lane>=16 -> K[16..31]
    const int bk = (lane >> 4) * 16;
    const int bc = (lane & 15);
    v16h bfrag[4];
#pragma unroll
    for (int j = 0; j < 4; j++)
      bfrag[j] = *(const v16h*)(Bs + (j * 16 + bc) * 32 + bk);
#pragma unroll
    for (int mi = 0; mi < 2; mi++)
#pragma unroll
      for (int j = 0; j < 4; j++)
        acc[mi][j] = __builtin_amdgcn_wmma_f32_16x16x32_f16(
            false, afrag[mi], false, bfrag[j], (short)0, acc[mi][j], false, false);
  }

  // D layout: lanes 0-15 -> N=lane, vgpr r -> M=r ; lanes 16-31 -> N=lane-16, M=r+8
  const int nBase = n0 + (lane & 15);
#pragma unroll
  for (int mi = 0; mi < 2; mi++) {
    const int mBase = m0 + wave * 32 + mi * 16 + ((lane >> 4) * 8);
#pragma unroll
    for (int j = 0; j < 4; j++) {
#pragma unroll
      for (int r = 0; r < 8; r++) {
        int gm = mBase + r;
        if (gm >= M) continue;
        int gn = nBase + j * 16;
        float v = acc[mi][j][r] + bias[gn];
        if (EPI == EPI_F16_RELU) v = v > 0.0f ? v : 0.0f;
        if (EPI == EPI_F16 || EPI == EPI_F16_RELU) {
          Ch[(size_t)gm * Nc + gn] = (_Float16)v;
        } else if (EPI == EPI_POOL) {
          atomicAdd(pool + (size_t)bids[gm] * Nc + gn, v);
        } else {
          Cf[(size_t)gm * Nc + gn] = v;
        }
      }
    }
  }
}

} // namespace

extern "C" void kernel_launch(void* const* d_in, const int* in_sizes, int n_in,
                              void* d_out, int out_size, void* d_ws, size_t ws_size,
                              hipStream_t stream) {
  (void)in_sizes; (void)n_in; (void)out_size; (void)ws_size;
  const int N = kN, E = kE, B = kB, H = kH, OUT = kOUT;
  const int NODE_DIMS[9] = {119, 9, 11, 12, 9, 5, 8, 2, 2};
  const int EDGE_DIMS[3] = {22, 6, 2};

  const int* x          = (const int*)d_in[0];
  const int* edge_attr  = (const int*)d_in[1];
  const int* edge_index = (const int*)d_in[2];
  const int* batch_ids  = (const int*)d_in[3];
  int p = 4;
  const float* node_tab[9]; for (int i = 0; i < 9; i++) node_tab[i] = (const float*)d_in[p++];
  const float* edge_tab[3]; for (int i = 0; i < 3; i++) edge_tab[i] = (const float*)d_in[p++];
  const float* Wnp = (const float*)d_in[p++];
  const float* bnp = (const float*)d_in[p++];
  const float* Wep = (const float*)d_in[p++];
  const float* bep = (const float*)d_in[p++];
  const float *W1[4], *b1[4], *W2[4], *b2[4], *gam[4], *bet[4];
  for (int l = 0; l < 4; l++) {
    W1[l]  = (const float*)d_in[p++]; b1[l]  = (const float*)d_in[p++];
    W2[l]  = (const float*)d_in[p++]; b2[l]  = (const float*)d_in[p++];
    gam[l] = (const float*)d_in[p++]; bet[l] = (const float*)d_in[p++];
  }
  const float* Wo1 = (const float*)d_in[p++];
  const float* bo1 = (const float*)d_in[p++];
  const float* Wo2 = (const float*)d_in[p++];
  const float* bo2 = (const float*)d_in[p++];

  // ---------------- workspace carve (aliased where lifetimes allow) ----------------
  char* ws = (char*)d_ws;
  size_t off = 0;
  auto carve = [&](size_t bytes) -> char* {
    char* r = ws + off;
    off = (off + bytes + 255) & ~(size_t)255;
    return r;
  };
  _Float16* wt_np = (_Float16*)carve((size_t)576 * 256 * 2);
  _Float16* wt_ep = (_Float16*)carve((size_t)96 * 256 * 2);
  _Float16* wt_l[8];
  for (int i = 0; i < 8; i++) wt_l[i] = (_Float16*)carve((size_t)256 * 256 * 2);
  _Float16* wt_o1 = (_Float16*)carve((size_t)256 * 256 * 2);
  _Float16* wt_o2 = (_Float16*)carve((size_t)256 * 512 * 2);
  _Float16* edge_h16 = (_Float16*)carve((size_t)E * H * 2);     // 409.6 MB, layer-invariant
  float* hbuf = (float*)carve((size_t)N * H * 4);
  char* Sbase = ws + off;                                        // 153.6 MB shared region
  float* agg  = (float*)carve((size_t)N * H * 4);
  float* zbuf = (float*)carve((size_t)N * H * 4);
  _Float16* zf16 = (_Float16*)carve((size_t)N * H * 2);
  _Float16* tf16 = (_Float16*)carve((size_t)N * H * 2);
  float* counts = (float*)carve((size_t)B * 4);
  _Float16* nodeEmb = (_Float16*)Sbase;  // N*576 f16, used before layer loop only
  _Float16* edgeEmb = (_Float16*)Sbase;  // E*96  f16, used before layer loop only

  auto blocks = [](size_t n, int bs) { return (unsigned)((n + (size_t)bs - 1) / bs); };

  // 1. weights -> transposed f16
  auto xp = [&](const float* W, _Float16* Wt, int K, int Nc) {
    wxpose<<<blocks((size_t)K * Nc, 256), 256, 0, stream>>>(W, Wt, K, Nc);
  };
  xp(Wnp, wt_np, 576, 256);
  xp(Wep, wt_ep, 96, 256);
  for (int l = 0; l < 4; l++) { xp(W1[l], wt_l[2 * l], 256, 256); xp(W2[l], wt_l[2 * l + 1], 256, 256); }
  xp(Wo1, wt_o1, 256, 256);
  xp(Wo2, wt_o2, 256, 512);

  auto gemm = [&](int epi, const _Float16* A, const _Float16* Bt, const float* bias,
                  float* Cf, _Float16* Ch, float* pool, int M, int K, int Nc) {
    dim3 grid(blocks((size_t)M, 128), (unsigned)(Nc / 64));
    dim3 blk(128);
    switch (epi) {
      case EPI_F32:      gemm16<EPI_F32     ><<<grid, blk, 0, stream>>>(A, Bt, bias, Cf, Ch, batch_ids, pool, M, K, Nc); break;
      case EPI_F16:      gemm16<EPI_F16     ><<<grid, blk, 0, stream>>>(A, Bt, bias, Cf, Ch, batch_ids, pool, M, K, Nc); break;
      case EPI_F16_RELU: gemm16<EPI_F16_RELU><<<grid, blk, 0, stream>>>(A, Bt, bias, Cf, Ch, batch_ids, pool, M, K, Nc); break;
      default:           gemm16<EPI_POOL    ><<<grid, blk, 0, stream>>>(A, Bt, bias, Cf, Ch, batch_ids, pool, M, K, Nc); break;
    }
  };

  // 2. node embedding -> projection -> h (f32)
  Tabs9 t9; for (int i = 0; i < 9; i++) { t9.t[i] = node_tab[i]; t9.d[i] = NODE_DIMS[i]; }
  embed_node<<<blocks((size_t)N * 576, 256), 256, 0, stream>>>(x, t9, nodeEmb, N);
  gemm(EPI_F32, nodeEmb, wt_np, bnp, hbuf, nullptr, nullptr, N, 576, 256);

  // 3. edge embedding -> projection -> edge_h (f16, reused by all layers)
  Tabs3 t3; for (int i = 0; i < 3; i++) { t3.t[i] = edge_tab[i]; t3.d[i] = EDGE_DIMS[i]; }
  embed_edge<<<blocks((size_t)E * 96, 256), 256, 0, stream>>>(edge_attr, t3, edgeEmb, E);
  gemm(EPI_F16, edgeEmb, wt_ep, bep, nullptr, edge_h16, nullptr, E, 96, 256);

  // 4. GINE layers
  const size_t nh = (size_t)N * H;
  for (int l = 0; l < 4; l++) {
    zero_f32<<<blocks(nh, 256), 256, 0, stream>>>(agg, nh);
    msg_agg<<<blocks((size_t)E * 32, 256), 256, 0, stream>>>(edge_index, edge_h16, hbuf, agg, E);
    add_convert<<<blocks(nh, 256), 256, 0, stream>>>(hbuf, agg, zbuf, zf16, nh);
    gemm(EPI_F16_RELU, zf16, wt_l[2 * l],     b1[l], nullptr, tf16, nullptr, N, 256, 256);
    gemm(EPI_F32,      tf16, wt_l[2 * l + 1], b2[l], zbuf,  nullptr, nullptr, N, 256, 256);
    ln_relu_res<<<blocks((size_t)N * 32, 256), 256, 0, stream>>>(zbuf, gam[l], bet[l], hbuf, N);
  }

  // 5. head + fused mean-pool
  f32_to_f16<<<blocks(nh, 256), 256, 0, stream>>>(hbuf, zf16, nh);
  gemm(EPI_F16_RELU, zf16, wt_o1, bo1, nullptr, tf16, nullptr, N, 256, 256);
  float* out = (float*)d_out;
  zero_f32<<<blocks((size_t)B * OUT, 256), 256, 0, stream>>>(out, (size_t)B * OUT);
  zero_f32<<<blocks((size_t)B, 256), 256, 0, stream>>>(counts, (size_t)B);
  gemm(EPI_POOL, tf16, wt_o2, bo2, nullptr, nullptr, out, N, 256, 512);
  count_nodes<<<blocks((size_t)N, 256), 256, 0, stream>>>(batch_ids, counts, N);
  pool_div<<<blocks((size_t)B * OUT, 256), 256, 0, stream>>>(out, counts, B * OUT);
}
